// DynamicGraphConstruction_42279658062323
// MI455X (gfx1250) — matrix operand order repaired and checked by reading, not compiled, not run on hardware
//
#include <hip/hip_runtime.h>
#include <hip/hip_bf16.h>

#define N_SRC 16384
#define N_DST 16384
#define DIM   128
#define KNN   16
#define E_TOT (N_SRC * KNN)
#define NT    (N_DST / 16)   // dst tiles
#define SCP   20             // score-tile row stride (floats): 16B aligned rows

typedef __attribute__((ext_vector_type(16))) __bf16 v16bf;
typedef __attribute__((ext_vector_type(8)))  float  v8f;

union BFrag  { v16bf v; uint4 q[2]; };
union F8Row  { uint4 q[2]; float f[8]; };

__device__ __forceinline__ unsigned short f2bf(float x) {
  unsigned u = __float_as_uint(x);
  unsigned r = u + 0x7fffu + ((u >> 16) & 1u);
  return (unsigned short)(r >> 16);
}
__device__ __forceinline__ float bf2f(unsigned short h) {
  return __uint_as_float(((unsigned)h) << 16);
}

// async 16-byte global -> LDS copy (ASYNCcnt-tracked, CDNA5)
__device__ __forceinline__ void async_copy16(const void* g, void* l) {
  const unsigned lds_addr = (unsigned)(size_t)l;   // LDS address: low 32 bits of the pointer
  asm volatile("global_load_async_to_lds_b128 %0, %1, off"
               :: "v"(lds_addr), "v"(g) : "memory");
}

// ---------------------------------------------------------------------------
// Kernel 1: split fp32 -> (bf16 hi, bf16 lo); per-dst-row h = 0.5*||d||^2
// ---------------------------------------------------------------------------
__global__ __launch_bounds__(128) void prep_kernel(
    const float* __restrict__ src, const float* __restrict__ dst,
    unsigned short* __restrict__ src_hi, unsigned short* __restrict__ src_lo,
    unsigned short* __restrict__ dst_hi, unsigned short* __restrict__ dst_lo,
    float* __restrict__ hbuf)
{
  __shared__ float red[128];
  const int b = blockIdx.x;
  const int t = threadIdx.x;
  if (b < N_DST) {
    const float x = dst[(size_t)b * DIM + t];
    const unsigned short h16 = f2bf(x);
    const unsigned short l16 = f2bf(x - bf2f(h16));
    dst_hi[(size_t)b * DIM + t] = h16;
    dst_lo[(size_t)b * DIM + t] = l16;
    red[t] = x * x;
    __syncthreads();
    for (int o = 64; o > 0; o >>= 1) {
      if (t < o) red[t] += red[t + o];
      __syncthreads();
    }
    if (t == 0) hbuf[b] = 0.5f * red[0];
  } else {
    const int r = b - N_DST;
    const float x = src[(size_t)r * DIM + t];
    const unsigned short h16 = f2bf(x);
    const unsigned short l16 = f2bf(x - bf2f(h16));
    src_hi[(size_t)r * DIM + t] = h16;
    src_lo[(size_t)r * DIM + t] = l16;
  }
}

// ---------------------------------------------------------------------------
// Kernel 2: fused bf16x2 WMMA GEMM + per-row running top-16
// B tiles staged via async global->LDS copies, double buffered.
// grid = N_SRC/64 blocks, 128 threads (4 waves, 16 src rows per wave)
// ---------------------------------------------------------------------------
__device__ __forceinline__ void issue_tile(
    const unsigned short* __restrict__ dst_hi,
    const unsigned short* __restrict__ dst_lo,
    int jbase, uint4* ldsbuf, int t)
{
#pragma unroll
  for (int i = 0; i < 2; ++i) {
    const int cid = t + i * 128;            // 0..255 chunk of 16B per matrix
    const int r = cid >> 4;                 // dst row within tile
    const int q = cid & 15;                 // 16B chunk within row
    const size_t goff = (size_t)(jbase + r) * DIM + q * 8;
    async_copy16(dst_hi + goff, ldsbuf + (r * 17 + q));             // hi region
    async_copy16(dst_lo + goff, ldsbuf + (16 * 17 + r * 17 + q));   // lo region
  }
}

__global__ __launch_bounds__(128) void gemm_topk_kernel(
    const unsigned short* __restrict__ src_hi, const unsigned short* __restrict__ src_lo,
    const unsigned short* __restrict__ dst_hi, const unsigned short* __restrict__ dst_lo,
    const float* __restrict__ hbuf,
    int* __restrict__ graph_src, int* __restrict__ graph_dst,
    float* __restrict__ dots)
{
  __shared__ uint4 btile[2][2][16][17];           // [buf][hi/lo][row][chunk]
  __shared__ alignas(16) float sc[4][16 * SCP];   // per-wave score tile, padded
  __shared__ float mv[4][32][KNN];                // per-lane top-k values (merge)
  __shared__ int   mi[4][32][KNN];                // per-lane top-k indices (merge)

  const int t    = threadIdx.x;
  const int lane = t & 31;
  const int wv   = t >> 5;
  const int m    = lane & 15;   // A: row in tile / B,C: column in tile
  const int hh   = lane >> 4;   // lane half (K/col-half selector)
  const int row  = blockIdx.x * 64 + wv * 16 + m;

  // --- preload A fragments (hi & lo) for this wave's 16 src rows ---
  BFrag a_hi[4], a_lo[4];
#pragma unroll
  for (int c = 0; c < 4; ++c) {
    const size_t base = (size_t)row * DIM + c * 32 + 16 * hh;
    a_hi[c].q[0] = *reinterpret_cast<const uint4*>(src_hi + base);
    a_hi[c].q[1] = *reinterpret_cast<const uint4*>(src_hi + base + 8);
    a_lo[c].q[0] = *reinterpret_cast<const uint4*>(src_lo + base);
    a_lo[c].q[1] = *reinterpret_cast<const uint4*>(src_lo + base + 8);
  }

  // --- register-resident sorted top-16 (descending), per lane ---
  float tv[KNN]; int ti[KNN];
#pragma unroll
  for (int s = 0; s < KNN; ++s) { tv[s] = -3.0e38f; ti[s] = 0; }

  // prologue: stage tile 0 into buffer 0
  issue_tile(dst_hi, dst_lo, 0, &btile[0][0][0][0], t);

  for (int jt = 0; jt < NT; ++jt) {
    const int cur = jt & 1;
    const int jbase = jt * 16;

    asm volatile("s_wait_asynccnt 0x0" ::: "memory");  // my copies landed
    __syncthreads();                                   // everyone's copies landed

    if (jt + 1 < NT)   // stage next tile into the other buffer
      issue_tile(dst_hi, dst_lo, jbase + 16, &btile[cur ^ 1][0][0][0], t);

    // --- bulk-load all 8 B fragments (distinct regs -> one wait, no stalls)
    // B frag: lane covers col n=m, K = c*32 + 16*hh + [0..16)
    const uint4* bb = &btile[cur][0][m][hh * 2];
    BFrag bh[4], bl[4];
#pragma unroll
    for (int c = 0; c < 4; ++c) {
      bh[c].q[0] = bb[c * 4];         bh[c].q[1] = bb[c * 4 + 1];
      bl[c].q[0] = bb[272 + c * 4];   bl[c].q[1] = bb[272 + c * 4 + 1];
    }

    // --- bf16x2 compensated GEMM: dot = sh*dh + sl*dh + sh*dl ---
    v8f acc = {};
#pragma unroll
    for (int c = 0; c < 4; ++c) {
      acc = __builtin_amdgcn_wmma_f32_16x16x32_bf16(false, a_hi[c].v, false, bh[c].v,
                                                    (short)0, acc, false, false);
      acc = __builtin_amdgcn_wmma_f32_16x16x32_bf16(false, a_lo[c].v, false, bh[c].v,
                                                    (short)0, acc, false, false);
      acc = __builtin_amdgcn_wmma_f32_16x16x32_bf16(false, a_hi[c].v, false, bl[c].v,
                                                    (short)0, acc, false, false);
    }

    // score = dot - 0.5*||d||^2 ; C layout: lane holds col N=m, rows M=v+8*hh
    const float hval = hbuf[jbase + m];
#pragma unroll
    for (int v = 0; v < 8; ++v)
      sc[wv][(v + 8 * hh) * SCP + m] = acc[v] - hval;
    asm volatile("s_wait_dscnt 0x0" ::: "memory");

    // --- scan: lane (m,hh) scans row m, cols [8*hh, 8*hh+8) ---
    F8Row rr;
    const uint4* rp = reinterpret_cast<const uint4*>(&sc[wv][m * SCP + hh * 8]);
    rr.q[0] = rp[0];
    rr.q[1] = rp[1];
#pragma unroll
    for (int j = 0; j < 8; ++j) {
      const float val = rr.f[j];
      if (val > tv[KNN - 1]) {              // strict > : lower index wins ties
        tv[KNN - 1] = val;
        ti[KNN - 1] = jbase + hh * 8 + j;
#pragma unroll
        for (int s = KNN - 1; s > 0; --s) { // stable bubble insert
          if (tv[s] > tv[s - 1]) {
            const float tf = tv[s]; tv[s] = tv[s - 1]; tv[s - 1] = tf;
            const int   tn = ti[s]; ti[s] = ti[s - 1]; ti[s - 1] = tn;
          }
        }
      }
    }
  }

  // --- merge the two half-lists per row, emit graph + dot values ---
#pragma unroll
  for (int s = 0; s < KNN; ++s) { mv[wv][lane][s] = tv[s]; mi[wv][lane][s] = ti[s]; }
  asm volatile("s_wait_dscnt 0x0" ::: "memory");

  if (lane < 16) {
    int pa = 0, pb = 0;
    for (int s = 0; s < KNN; ++s) {
      const float va = mv[wv][lane][pa];      const int ia = mi[wv][lane][pa];
      const float vb = mv[wv][lane + 16][pb]; const int ib = mi[wv][lane + 16][pb];
      const bool takeA = (va > vb) || (va == vb && ia < ib);
      const float v = takeA ? va : vb;
      const int   i = takeA ? ia : ib;
      if (takeA) ++pa; else ++pb;
      const int e = row * KNN + s;
      graph_src[e] = row;
      graph_dst[e] = i;
      dots[e] = v + hbuf[i];                 // likelihood = score + 0.5*||d||^2
    }
  }
}

// ---------------------------------------------------------------------------
// Kernel 3a: per-block partial (sum, sumsq) over 512 likelihoods
// ---------------------------------------------------------------------------
__global__ __launch_bounds__(256) void bn_partial_kernel(
    const float* __restrict__ dots, float* __restrict__ partials)
{
  __shared__ float s1[256], s2[256];
  const int t = threadIdx.x, b = blockIdx.x;
  const float x0 = dots[b * 512 + t];
  const float x1 = dots[b * 512 + 256 + t];
  s1[t] = x0 + x1;
  s2[t] = x0 * x0 + x1 * x1;
  __syncthreads();
  for (int o = 128; o > 0; o >>= 1) {
    if (t < o) { s1[t] += s1[t + o]; s2[t] += s2[t + o]; }
    __syncthreads();
  }
  if (t == 0) { partials[b * 2] = s1[0]; partials[b * 2 + 1] = s2[0]; }
}

// ---------------------------------------------------------------------------
// Kernel 3b: reduce 512 partials -> stats[0]=mean, stats[1]=var (biased)
// ---------------------------------------------------------------------------
__global__ __launch_bounds__(256) void bn_stats_kernel(
    const float* __restrict__ partials, float* __restrict__ stats)
{
  __shared__ float s1[256], s2[256];
  const int t = threadIdx.x;
  s1[t] = partials[t * 2] + partials[(t + 256) * 2];
  s2[t] = partials[t * 2 + 1] + partials[(t + 256) * 2 + 1];
  __syncthreads();
  for (int o = 128; o > 0; o >>= 1) {
    if (t < o) { s1[t] += s1[t + o]; s2[t] += s2[t + o]; }
    __syncthreads();
  }
  if (t == 0) {
    const float inv = 1.0f / (float)E_TOT;
    const float mean = s1[0] * inv;
    stats[0] = mean;
    stats[1] = s2[0] * inv - mean * mean;
  }
}

// ---------------------------------------------------------------------------
// Kernel 4: BN -> exp -> local per-row (16-edge) degree normalization
// ---------------------------------------------------------------------------
__global__ __launch_bounds__(256) void edge_weight_kernel(
    const float* __restrict__ dots, const float* __restrict__ stats,
    const float* __restrict__ gamma, const float* __restrict__ beta,
    float* __restrict__ w_out)
{
  const int e = blockIdx.x * 256 + threadIdx.x;
  const float rstd = rsqrtf(stats[1] + 1e-5f);
  const float logit = (dots[e] - stats[0]) * rstd * gamma[0] + beta[0];
  const float w = expf(logit);
  float deg = w;
#pragma unroll
  for (int msk = 1; msk < 16; msk <<= 1)
    deg += __shfl_xor(deg, msk, 16);   // sum over this src row's 16 edges
  w_out[e] = w / (1e-12f + deg);
}

// ---------------------------------------------------------------------------
extern "C" void kernel_launch(void* const* d_in, const int* in_sizes, int n_in,
                              void* d_out, int out_size, void* d_ws, size_t ws_size,
                              hipStream_t stream) {
  const float* src   = (const float*)d_in[0];
  const float* dst   = (const float*)d_in[1];
  const float* gamma = (const float*)d_in[2];
  const float* beta  = (const float*)d_in[3];

  char* ws = (char*)d_ws;
  const size_t MAT = (size_t)N_DST * DIM * 2;   // 4 MiB per bf16 matrix
  unsigned short* dst_hi = (unsigned short*)(ws);
  unsigned short* dst_lo = (unsigned short*)(ws + MAT);
  unsigned short* src_hi = (unsigned short*)(ws + 2 * MAT);
  unsigned short* src_lo = (unsigned short*)(ws + 3 * MAT);
  float* hbuf     = (float*)(ws + 4 * MAT);
  float* dots     = (float*)(ws + 4 * MAT + (size_t)N_DST * 4);
  float* partials = (float*)(ws + 4 * MAT + (size_t)N_DST * 4 + (size_t)E_TOT * 4);
  float* stats    = partials + 1024;

  int*   graph_out = (int*)d_out;                 // [2, N*K] flat (int32 bits)
  float* w_out     = (float*)d_out + 2 * E_TOT;   // [N*K] weights

  prep_kernel<<<dim3(N_DST + N_SRC), dim3(128), 0, stream>>>(
      src, dst, src_hi, src_lo, dst_hi, dst_lo, hbuf);

  gemm_topk_kernel<<<dim3(N_SRC / 64), dim3(128), 0, stream>>>(
      src_hi, src_lo, dst_hi, dst_lo, hbuf,
      graph_out, graph_out + E_TOT, dots);

  bn_partial_kernel<<<dim3(512), dim3(256), 0, stream>>>(dots, partials);
  bn_stats_kernel<<<dim3(1), dim3(256), 0, stream>>>(partials, stats);

  edge_weight_kernel<<<dim3(E_TOT / 256), dim3(256), 0, stream>>>(
      dots, stats, gamma, beta, w_out);
}
